// BEVEncoderLayer_81046032875793
// MI455X (gfx1250) — matrix operand order repaired and checked by reading, not compiled, not run on hardware
//
#include <hip/hip_runtime.h>
#include <math.h>

#define HW_  2500
#define B_   2
#define C_   256
#define NH_  8
#define DH_  64
#define NC_  6
#define HF_  64
#define WF_  176
#define DFF_ 1024
#define MTOK (HW_ * B_)

typedef __attribute__((ext_vector_type(16))) __bf16 bf16x16;
typedef __attribute__((ext_vector_type(8)))  float  f32x8;

union FragAB { bf16x16 v; unsigned u[8]; __bf16 h[16]; };

__device__ __forceinline__ __bf16 f2bf(float f) { return (__bf16)f; }

// A-operand k mapping (contiguous, matches B): half 0 -> K 0..15 ; half 1 -> K 16..31
__device__ __forceinline__ int kmap(int i, int half) {
    return 2 * i + half * 16;
}

// ---------------------------------------------------------------------------
// Generic GEMM: C[M,N] = A[M,K] @ op(W) + bias, op = W^T (N,K row-major) if
// TRANS_B else W (K,N row-major). bf16 WMMA, f32 accumulate. Optional GELU.
// Block tile 128x64, 8 waves of 32x32, BK=32. global_prefetch on next K tile.
// ---------------------------------------------------------------------------
template <bool TRANS_B, bool GELU>
__global__ __launch_bounds__(256) void gemm_bf16(const float* __restrict__ A,
                                                 const float* __restrict__ W,
                                                 const float* __restrict__ bias,
                                                 float* __restrict__ Cout,
                                                 int M, int N, int K) {
    __shared__ __bf16 As[128][34];   // [m][k]
    __shared__ __bf16 Bs[64][34];    // [n][k]  (k-contiguous per column)
    const int tid = threadIdx.x;
    const int lane = tid & 31, wave = tid >> 5;
    const int half = lane >> 4, l15 = lane & 15;
    const int wm = wave & 3, wn = wave >> 2;
    const int m0 = blockIdx.x * 128, n0 = blockIdx.y * 64;

    f32x8 acc[2][2] = {};
    for (int k0 = 0; k0 < K; k0 += 32) {
#pragma unroll
        for (int i = 0; i < 16; ++i) {              // 128x32 A tile
            int e = i * 256 + tid;
            int r = e >> 5, c = e & 31;
            int gr = m0 + r;
            float v = (gr < M) ? A[(size_t)gr * K + k0 + c] : 0.f;
            As[r][c] = f2bf(v);
        }
#pragma unroll
        for (int i = 0; i < 8; ++i) {               // 32x64 B tile
            int e = i * 256 + tid;
            int kl = e & 31, nl = e >> 5;
            int gn = n0 + nl, gk = k0 + kl;
            float v = TRANS_B ? W[(size_t)gn * K + gk] : W[(size_t)gk * N + gn];
            Bs[nl][kl] = f2bf(v);
        }
        // prefetch next K tile while WMMAs run (global_prefetch_b8)
        if (k0 + 32 < K) {
            int gr = m0 + (tid >> 1);               // 2 threads per row, 32-col tile
            if (gr < M) __builtin_prefetch(&A[(size_t)gr * K + k0 + 32 + (tid & 1) * 16], 0, 3);
            int gn = n0 + (tid & 63);
            int gk = k0 + 32 + ((tid >> 6) << 3);
            __builtin_prefetch(TRANS_B ? &W[(size_t)gn * K + gk] : &W[(size_t)gk * N + gn], 0, 3);
        }
        __syncthreads();

        FragAB a[2], b[2];
#pragma unroll
        for (int mi = 0; mi < 2; ++mi) {
            int row = wm * 32 + mi * 16 + l15;
#pragma unroll
            for (int i = 0; i < 8; ++i)
                a[mi].u[i] = *(const unsigned*)&As[row][kmap(i, half)];
        }
#pragma unroll
        for (int ni = 0; ni < 2; ++ni) {
            int col = wn * 32 + ni * 16 + l15;
            int kb = half * 16;
#pragma unroll
            for (int i = 0; i < 8; ++i)
                b[ni].u[i] = *(const unsigned*)&Bs[col][kb + 2 * i];
        }
#pragma unroll
        for (int mi = 0; mi < 2; ++mi)
#pragma unroll
            for (int ni = 0; ni < 2; ++ni)
                acc[mi][ni] = __builtin_amdgcn_wmma_f32_16x16x32_bf16(
                    false, a[mi].v, false, b[ni].v, (short)0, acc[mi][ni], false, false);
        __syncthreads();
    }

#pragma unroll
    for (int mi = 0; mi < 2; ++mi) {
#pragma unroll
        for (int ni = 0; ni < 2; ++ni) {
            int col = n0 + wn * 32 + ni * 16 + l15;
            float bv = bias ? bias[col] : 0.f;
#pragma unroll
            for (int v = 0; v < 8; ++v) {
                int row = m0 + wm * 32 + mi * 16 + half * 8 + v;
                if (row < M) {
                    float x = acc[mi][ni][v] + bv;
                    if (GELU) x = 0.5f * x * (1.f + erff(x * 0.70710678118f));
                    Cout[(size_t)row * N + col] = x;
                }
            }
        }
    }
}

// ---------------------------------------------------------------------------
// Fused multi-head attention (flash style). One block per (b, h, 64-row tile),
// 4 waves x 16 rows. K/V staged in LDS in 64-col chunks; online softmax.
// Q/K/V layout: [token = l*B_+b, h*64+d], f32.
// ---------------------------------------------------------------------------
__global__ __launch_bounds__(128) void attn_fwd(const float* __restrict__ Q,
                                                const float* __restrict__ Kp,
                                                const float* __restrict__ Vp,
                                                float* __restrict__ O,
                                                int L, int S) {
    __shared__ __bf16 Ks[64][66];        // [s][d]
    __shared__ __bf16 Vs[64][66];        // [d][s]
    __shared__ __bf16 Ps[4][16][66];     // per-wave P scratch [m][s]
    const int tid = threadIdx.x;
    const int lane = tid & 31, wave = tid >> 5;
    const int half = lane >> 4, l15 = lane & 15;
    const int b = blockIdx.z, h = blockIdx.y;
    const int m0 = blockIdx.x * 64 + wave * 16;
    const int ch = h * DH_;
    const float scale = 0.125f;          // 1/sqrt(64)

    FragAB qa[2];                         // Q tile 16x64, scale folded in
#pragma unroll
    for (int ks = 0; ks < 2; ++ks) {
        int row = m0 + l15;
        const float* qrow = Q + (size_t)(row * B_ + b) * C_ + ch + ks * 32;
#pragma unroll
        for (int i = 0; i < 8; ++i) {
            int kb = kmap(i, half);
            float v0 = 0.f, v1 = 0.f;
            if (row < L) { v0 = qrow[kb] * scale; v1 = qrow[kb + 1] * scale; }
            qa[ks].h[2 * i] = f2bf(v0);
            qa[ks].h[2 * i + 1] = f2bf(v1);
        }
    }

    float m_i[8], l_i[8];
#pragma unroll
    for (int v = 0; v < 8; ++v) { m_i[v] = -1e30f; l_i[v] = 0.f; }
    f32x8 o[4] = {};

    for (int sc = 0; sc < S; sc += 64) {
        for (int idx = tid; idx < 64 * 64; idx += 128) {
            int sl = idx >> 6, d = idx & 63;
            int gs = sc + sl;
            float kv = 0.f, vv = 0.f;
            if (gs < S) {
                size_t base = (size_t)(gs * B_ + b) * C_ + ch;
                kv = Kp[base + d];
                vv = Vp[base + d];
            }
            Ks[sl][d] = f2bf(kv);
            Vs[d][sl] = f2bf(vv);
        }
        // prefetch next chunk (global_prefetch_b8)
        {
            int gs = sc + 64 + (tid >> 1);
            if (gs < S) {
                size_t base = (size_t)(gs * B_ + b) * C_ + ch + (tid & 1) * 32;
                __builtin_prefetch(&Kp[base], 0, 3);
                __builtin_prefetch(&Vp[base], 0, 3);
            }
        }
        __syncthreads();

        f32x8 s[4] = {};
#pragma unroll
        for (int nt = 0; nt < 4; ++nt) {
#pragma unroll
            for (int ks = 0; ks < 2; ++ks) {
                FragAB bfr;
                int scol = nt * 16 + l15;
                int kb = ks * 32 + half * 16;
#pragma unroll
                for (int i = 0; i < 8; ++i)
                    bfr.u[i] = *(const unsigned*)&Ks[scol][kb + 2 * i];
                s[nt] = __builtin_amdgcn_wmma_f32_16x16x32_bf16(
                    false, qa[ks].v, false, bfr.v, (short)0, s[nt], false, false);
            }
            int gcol = sc + nt * 16 + l15;
            if (gcol >= S) {
#pragma unroll
                for (int v = 0; v < 8; ++v) s[nt][v] = -1e30f;
            }
        }

        // online softmax: row = half*8+v, its 16 cols live across one 16-lane half
        float rs[8];
#pragma unroll
        for (int v = 0; v < 8; ++v) {
            float t = fmaxf(fmaxf(s[0][v], s[1][v]), fmaxf(s[2][v], s[3][v]));
#pragma unroll
            for (int mask = 1; mask < 16; mask <<= 1)
                t = fmaxf(t, __shfl_xor(t, mask, 16));
            float mn = fmaxf(m_i[v], t);
            float corr = __expf(m_i[v] - mn);
            m_i[v] = mn;
            l_i[v] *= corr;
#pragma unroll
            for (int nt = 0; nt < 4; ++nt) o[nt][v] *= corr;
            rs[v] = 0.f;
        }
#pragma unroll
        for (int nt = 0; nt < 4; ++nt) {
#pragma unroll
            for (int v = 0; v < 8; ++v) {
                float p = __expf(s[nt][v] - m_i[v]);
                rs[v] += p;
                Ps[wave][half * 8 + v][nt * 16 + l15] = f2bf(p);
            }
        }
#pragma unroll
        for (int v = 0; v < 8; ++v) {
            float t = rs[v];
#pragma unroll
            for (int mask = 1; mask < 16; mask <<= 1)
                t += __shfl_xor(t, mask, 16);
            l_i[v] += t;
        }

        // P @ V  (within-wave LDS ordering covers Ps store->load)
        FragAB pa[2];
#pragma unroll
        for (int ks = 0; ks < 2; ++ks) {
#pragma unroll
            for (int i = 0; i < 8; ++i)
                pa[ks].u[i] = *(const unsigned*)&Ps[wave][l15][ks * 32 + kmap(i, half)];
        }
#pragma unroll
        for (int nt = 0; nt < 4; ++nt) {
#pragma unroll
            for (int ks = 0; ks < 2; ++ks) {
                FragAB bfr;
                int dcol = nt * 16 + l15;
                int kb = ks * 32 + half * 16;
#pragma unroll
                for (int i = 0; i < 8; ++i)
                    bfr.u[i] = *(const unsigned*)&Vs[dcol][kb + 2 * i];
                o[nt] = __builtin_amdgcn_wmma_f32_16x16x32_bf16(
                    false, pa[ks].v, false, bfr.v, (short)0, o[nt], false, false);
            }
        }
        __syncthreads();
    }

#pragma unroll
    for (int nt = 0; nt < 4; ++nt) {
#pragma unroll
        for (int v = 0; v < 8; ++v) {
            int row = m0 + half * 8 + v;
            if (row < L)
                O[(size_t)(row * B_ + b) * C_ + ch + nt * 16 + l15] = o[nt][v] / l_i[v];
        }
    }
}

// ---------------------------------------------------------------------------
// out = LayerNorm(x + res) * g + b   (one block per token, C_=256 threads)
// ---------------------------------------------------------------------------
__global__ __launch_bounds__(256) void add_ln(const float* __restrict__ x,
                                              const float* __restrict__ res,
                                              const float* __restrict__ g,
                                              const float* __restrict__ bta,
                                              float* __restrict__ out, int M) {
    __shared__ float red[256];
    int m = blockIdx.x, c = threadIdx.x;
    float t = x[(size_t)m * C_ + c] + res[(size_t)m * C_ + c];
    red[c] = t;
    __syncthreads();
    for (int s = 128; s > 0; s >>= 1) {
        if (c < s) red[c] += red[c + s];
        __syncthreads();
    }
    float mean = red[0] * (1.f / C_);
    __syncthreads();
    float d = t - mean;
    red[c] = d * d;
    __syncthreads();
    for (int s = 128; s > 0; s >>= 1) {
        if (c < s) red[c] += red[c + s];
        __syncthreads();
    }
    float var = red[0] * (1.f / C_);
    out[(size_t)m * C_ + c] = d * rsqrtf(var + 1e-5f) * g[c] + bta[c];
}

// ---------------------------------------------------------------------------
// Camera projection + bilinear grid-sample (zeros pad), mean over cameras.
// grid->pixel collapses to x = u - 0.5, y = v - 0.5.
// out[(q*B_+b)*C_ + c]
// ---------------------------------------------------------------------------
__global__ __launch_bounds__(256) void cam_sample(const float* __restrict__ feat,
                                                  const float* __restrict__ refp,
                                                  const float* __restrict__ intr,
                                                  const float* __restrict__ extr,
                                                  float* __restrict__ out) {
    int q = blockIdx.x, b = blockIdx.y, c = threadIdx.x;
    float rx = refp[q * 3 + 0], ry = refp[q * 3 + 1], rz = refp[q * 3 + 2];
    float acc = 0.f;
#pragma unroll
    for (int cam = 0; cam < NC_; ++cam) {
        const float* E = extr + (size_t)(b * NC_ + cam) * 16;
        const float* Km = intr + (size_t)(b * NC_ + cam) * 9;
        float dx = rx - E[3], dy = ry - E[7], dz = rz - E[11];
        float pc0 = E[0] * dx + E[4] * dy + E[8] * dz;    // R^T (ref - t)
        float pc1 = E[1] * dx + E[5] * dy + E[9] * dz;
        float pc2 = E[2] * dx + E[6] * dy + E[10] * dz;
        float i0 = Km[0] * pc0 + Km[1] * pc1 + Km[2] * pc2;
        float i1 = Km[3] * pc0 + Km[4] * pc1 + Km[5] * pc2;
        float i2 = Km[6] * pc0 + Km[7] * pc1 + Km[8] * pc2;
        float z = fmaxf(i2, 1e-5f);
        float x = i0 / z - 0.5f, y = i1 / z - 0.5f;
        float x0f = floorf(x), y0f = floorf(y);
        float wx = x - x0f, wy = y - y0f;
        int x0 = (int)x0f, y0 = (int)y0f;
        const float* F = feat + ((size_t)(b * NC_ + cam) * C_ + c) * (HF_ * WF_);
        float s = 0.f;
#pragma unroll
        for (int dyi = 0; dyi < 2; ++dyi)
#pragma unroll
            for (int dxi = 0; dxi < 2; ++dxi) {
                int xi = x0 + dxi, yi = y0 + dyi;
                float w = (dxi ? wx : 1.f - wx) * (dyi ? wy : 1.f - wy);
                if (xi >= 0 && xi < WF_ && yi >= 0 && yi < HF_)
                    s += F[yi * WF_ + xi] * w;
            }
        acc += s;
    }
    out[((size_t)q * B_ + b) * C_ + c] = acc * (1.f / NC_);
}

extern "C" void kernel_launch(void* const* d_in, const int* in_sizes, int n_in,
                              void* d_out, int out_size, void* d_ws, size_t ws_size,
                              hipStream_t stream) {
    (void)in_sizes; (void)n_in; (void)out_size; (void)ws_size;
    const float* bev  = (const float*)d_in[0];
    const float* prev = (const float*)d_in[1];
    const float* feat = (const float*)d_in[2];
    const float* refp = (const float*)d_in[3];
    const float* intr = (const float*)d_in[4];
    const float* extr = (const float*)d_in[5];
    const float* in_w = (const float*)d_in[7];
    const float* in_b = (const float*)d_in[8];
    const float* ow   = (const float*)d_in[9];
    const float* ob   = (const float*)d_in[10];
    const float* tsg  = (const float*)d_in[11];
    const float* tsb  = (const float*)d_in[12];
    const float* scw  = (const float*)d_in[13];
    const float* scb  = (const float*)d_in[14];
    const float* sng  = (const float*)d_in[15];
    const float* snb  = (const float*)d_in[16];
    const float* w1   = (const float*)d_in[17];
    const float* b1   = (const float*)d_in[18];
    const float* w2   = (const float*)d_in[19];
    const float* b2   = (const float*)d_in[20];
    const float* fng  = (const float*)d_in[21];
    const float* fnb  = (const float*)d_in[22];

    float* ws = (float*)d_ws;
    const size_t SLOT = (size_t)MTOK * C_;  // 1.28M floats per slot
    float* q    = ws + 0 * SLOT;
    float* k    = ws + 1 * SLOT;
    float* v    = ws + 2 * SLOT;
    float* attn = ws + 3 * SLOT;
    float* tsa  = ws + 0 * SLOT;  // reuse: q dead after attention
    float* x1   = ws + 1 * SLOT;  // reuse: k dead
    float* samp = ws + 2 * SLOT;  // reuse: v dead
    float* sca  = ws + 3 * SLOT;  // reuse: attn dead after out-proj
    float* x2   = ws + 0 * SLOT;  // reuse: tsa dead after LN1
    float* hbuf = ws + 4 * SLOT;  // 4 slots (5000 x 1024)
    float* ffo  = ws + 1 * SLOT;  // reuse: x1 dead after LN2

    dim3 blk(256);
    dim3 g1((MTOK + 127) / 128, C_ / 64);

    // QKV projections (x @ W^T + b)
    gemm_bf16<true, false><<<g1, blk, 0, stream>>>(bev,  in_w,              in_b,          q, MTOK, C_, C_);
    gemm_bf16<true, false><<<g1, blk, 0, stream>>>(prev, in_w + C_ * C_,    in_b + C_,     k, MTOK, C_, C_);
    gemm_bf16<true, false><<<g1, blk, 0, stream>>>(prev, in_w + 2 * C_ * C_, in_b + 2 * C_, v, MTOK, C_, C_);

    // fused attention
    dim3 ga((HW_ + 63) / 64, NH_, B_);
    attn_fwd<<<ga, dim3(128), 0, stream>>>(q, k, v, attn, HW_, HW_);

    // out projection + LN1
    gemm_bf16<true, false><<<g1, blk, 0, stream>>>(attn, ow, ob, tsa, MTOK, C_, C_);
    add_ln<<<dim3(MTOK), blk, 0, stream>>>(bev, tsa, tsg, tsb, x1, MTOK);

    // spatial cross attention: camera sample + projection + LN2
    cam_sample<<<dim3(HW_, B_), blk, 0, stream>>>(feat, refp, intr, extr, samp);
    gemm_bf16<true, false><<<g1, blk, 0, stream>>>(samp, scw, scb, sca, MTOK, C_, C_);
    add_ln<<<dim3(MTOK), blk, 0, stream>>>(x1, sca, sng, snb, x2, MTOK);

    // FFN + LN3
    dim3 g2((MTOK + 127) / 128, DFF_ / 64);
    gemm_bf16<false, true><<<g2, blk, 0, stream>>>(x2, w1, b1, hbuf, MTOK, DFF_, C_);
    gemm_bf16<false, false><<<g1, blk, 0, stream>>>(hbuf, w2, b2, ffo, MTOK, C_, DFF_);
    add_ln<<<dim3(MTOK), blk, 0, stream>>>(x2, ffo, fng, fnb, (float*)d_out, MTOK);
}
